// FNC_BERTopicModel_withDot_22024592293917
// MI455X (gfx1250) — matrix operand order repaired and checked by reading, not compiled, not run on hardware
//
#include <hip/hip_runtime.h>

typedef __bf16  v16bf __attribute__((ext_vector_type(16)));
typedef float   v8f   __attribute__((ext_vector_type(8)));

#define EHID   256
#define E_IN   300
#define EPAD   320
#define BATCH  256
#define TTOT   512
#define WHEAD  64
#define WBODY  448
#define BROWS  32
#define MLPN   512

#ifndef __has_builtin
#define __has_builtin(x) 0
#endif
#if __has_builtin(__builtin_amdgcn_tensor_load_to_lds) && \
    __has_builtin(__builtin_amdgcn_s_wait_tensorcnt)
#define USE_TDM 1
#else
#define USE_TDM 0
#endif

// ---------- helpers ----------
__device__ __forceinline__ unsigned short f2bf(float f) {
  unsigned u = __float_as_uint(f);
  u += 0x7FFFu + ((u >> 16) & 1u);          // round-to-nearest-even
  return (unsigned short)(u >> 16);
}
__device__ __forceinline__ float sigm(float x) {
  return 1.0f / (1.0f + __expf(-x));
}
__device__ __forceinline__ float tanh_fast(float x) {
  return 1.0f - 2.0f / (__expf(2.0f * x) + 1.0f);
}

struct Params {
  const float* wih[4];
  const float* whh[4];
  const float* bih[4];
  const float* bhh[4];
};

#define WIH_TOT (4 * 1024 * EPAD)
#define WHH_TOT (4 * 1024 * EHID)
#define XP_TOT  ((size_t)BATCH * TTOT * EPAD)

#if USE_TDM
// Issue a 2D TDM load: BROWS rows x (EPAD bf16) from global (row stride
// TTOT*EPAD elems) into contiguous LDS at lds_byte_addr. D# per ISA ch.8.
// 6-arg builtin form: (v4u g0, v8i g1, v4i g2, v4i g3, v8i g4, i32 cpol).
__device__ __forceinline__ void tdm_load_x(unsigned lds_byte_addr,
                                           const unsigned short* gptr) {
  typedef unsigned v4u __attribute__((ext_vector_type(4)));
  typedef int      v8i __attribute__((ext_vector_type(8)));
  typedef int      v4i __attribute__((ext_vector_type(4)));
  const unsigned long long ga = (unsigned long long)gptr;
  const unsigned dim0   = EPAD / 2;                  // row length in 4B units = 160
  const unsigned rows   = BROWS;                     // 32
  const unsigned stride = (TTOT * EPAD) / 2;         // 81920 (4B units)
  v4u g0;
  g0[0] = 1u;                                        // count=1, user desc
  g0[1] = lds_byte_addr;                             // lds_addr
  g0[2] = (unsigned)(ga & 0xFFFFFFFFu);              // global_addr[31:0]
  g0[3] = (unsigned)((ga >> 32) & 0x1FFFFFFu) | (2u << 30);  // addr[56:32] | type=2
  v8i g1;
  g1[0] = (int)(2u << 16);                           // data_size=2 (4 bytes)
  g1[1] = (int)(dim0 << 16);                         // tensor_dim0[15:0]
  g1[2] = (int)(rows << 16);                         // dim0 hi=0 | tensor_dim1 lo
  g1[3] = (int)(dim0 << 16);                         // dim1 hi=0 | tile_dim0
  g1[4] = (int)rows;                                 // tile_dim1 | tile_dim2=0
  g1[5] = (int)stride;                               // tensor_dim0_stride lo32
  g1[6] = 0;                                         // stride0 hi | stride1 lo
  g1[7] = 0;
  v4i z4 = {0, 0, 0, 0};
  v8i z8 = {0, 0, 0, 0, 0, 0, 0, 0};
  __builtin_amdgcn_tensor_load_to_lds(g0, g1, z4, z4, z8, 0);
}
#endif

// ---------- kernel 0: bf16 conversion (x padded, weights), bias sums ----------
__global__ void convert_kernel(const float* __restrict__ x, Params p,
                               unsigned short* __restrict__ xb,
                               unsigned short* __restrict__ wihp,
                               unsigned short* __restrict__ whhb,
                               float* __restrict__ bias) {
  size_t idx = (size_t)blockIdx.x * 256 + threadIdx.x;
  if (idx < XP_TOT) {
    size_t bt = idx / EPAD;
    int e = (int)(idx - bt * EPAD);
    float v = (e < E_IN) ? x[bt * E_IN + e] : 0.0f;
    xb[idx] = f2bf(v);
    return;
  }
  idx -= XP_TOT;
  if (idx < WIH_TOT) {
    int d = (int)(idx / (1024 * EPAD));
    int r = (int)(idx - (size_t)d * (1024 * EPAD));
    int j = r / EPAD, e = r - j * EPAD;
    float v = (e < E_IN) ? p.wih[d][(size_t)j * E_IN + e] : 0.0f;
    wihp[idx] = f2bf(v);
    return;
  }
  idx -= WIH_TOT;
  if (idx < WHH_TOT) {
    int d = (int)(idx / (1024 * EHID));
    int r = (int)(idx - (size_t)d * (1024 * EHID));
    whhb[idx] = f2bf(p.whh[d][r]);
    return;
  }
  idx -= WHH_TOT;
  if (idx < 4096) {
    int d = (int)(idx / 1024), j = (int)(idx - (size_t)d * 1024);
    bias[idx] = p.bih[d][j] + p.bhh[d][j];
  }
}

// ---------- kernel 1: persistent BiLSTM recurrence, WMMA bf16 ----------
// grid = (4 directions, 8 batch-chunks of 32 rows), block = 256 (8 waves).
// Wave w covers hblks {w, w+8}; within a K-chunk each B fragment is shared
// by two M-blocks (batch rows 0-15 / 16-31): 8 WMMAs per 4 B loads.
// Double-buffered LDS; x tile staged by the Tensor Data Mover (wave 0).
__global__ void __launch_bounds__(256) bilstm_kernel(
    const unsigned short* __restrict__ xb,
    const unsigned short* __restrict__ wihp,
    const unsigned short* __restrict__ whhb,
    const float* __restrict__ bias,
    float* __restrict__ hsum) {
  const int d    = blockIdx.x;
  const int b0   = blockIdx.y * BROWS;
  const int T    = (d < 2) ? WHEAD : WBODY;
  const int toff = (d < 2) ? 0 : WHEAD;
  const bool rev = (d & 1);

  // single allocation so LDS byte offsets are well-defined (TDM lds_addr)
  __shared__ unsigned short smem[2 * BROWS * EPAD + 2 * BROWS * EHID];
  typedef unsigned short XTile[BROWS][EPAD];
  typedef unsigned short HTile[BROWS][EHID];
  XTile* xt = (XTile*)smem;                          // [2][32][320]
  HTile* ht = (HTile*)(smem + 2 * BROWS * EPAD);     // [2][32][256]
#define XT_OFF(buf) ((unsigned)((buf) * BROWS * EPAD * 2))

  const int tid   = threadIdx.x;
  const int lane  = tid & 31;
  const int w     = tid >> 5;                  // wave 0..7
  const int ln15  = lane & 15;
  const int hi16  = (lane >> 4);
  const int akoff = hi16 << 4;                 // A frag: K +0 / +16
  const int bkoff = hi16 << 4;                 // B frag: K +0 / +16
  const int rbase = hi16 << 3;                 // C/D row base: 0 / 8

  const unsigned short* wih_d = wihp + (size_t)d * 1024 * EPAD;
  const unsigned short* whh_d = whhb + (size_t)d * 1024 * EHID;
  const float* bias_d = bias + d * 1024;

  // zero h state (buffer 0)
  for (int i = tid; i < BROWS * EHID; i += 256)
    ((unsigned short*)ht[0])[i] = 0;
  // stage first x tile into buffer 0
  {
    const int tg = toff + (rev ? (T - 1) : 0);
    const unsigned short* src = &xb[((size_t)b0 * TTOT + tg) * EPAD];
#if USE_TDM
    if (w == 0) tdm_load_x(XT_OFF(0), src);
#else
    for (int i = tid; i < BROWS * (EPAD / 8); i += 256) {
      int r = i / (EPAD / 8), q = i - r * (EPAD / 8);
      *(uint4*)&xt[0][r][q * 8] =
          *(const uint4*)&src[(size_t)r * TTOT * EPAD + q * 8];
    }
#endif
  }

  // loop-invariant per-lane biases, indexed [hg]
  float bi[2], bf_[2], bg_[2], bo_[2];
#pragma unroll
  for (int hg = 0; hg < 2; ++hg) {
    const int jb = (w + hg * 8) * 16 + ln15;
    bi[hg]  = bias_d[jb];        bf_[hg] = bias_d[256 + jb];
    bg_[hg] = bias_d[512 + jb];  bo_[hg] = bias_d[768 + jb];
  }

  v8f cc[2][2] = {}, ss[2][2] = {};   // [hg][mb]

  for (int tt = 0; tt < T; ++tt) {
    const int cur = tt & 1, nxt = cur ^ 1;
#if USE_TDM
    if (w == 0) __builtin_amdgcn_s_wait_tensorcnt(0);  // x[cur] landed
#endif
    __syncthreads();

    // stage next step's x tile into the other buffer
    if (tt + 1 < T) {
      const int tg = toff + (rev ? (T - 2 - tt) : (tt + 1));
      const unsigned short* src = &xb[((size_t)b0 * TTOT + tg) * EPAD];
#if USE_TDM
      if (w == 0) tdm_load_x(XT_OFF(nxt), src);
#else
      for (int i = tid; i < BROWS * (EPAD / 8); i += 256) {
        int r = i / (EPAD / 8), q = i - r * (EPAD / 8);
        *(uint4*)&xt[nxt][r][q * 8] =
            *(const uint4*)&src[(size_t)r * TTOT * EPAD + q * 8];
      }
#endif
    }
    // prefetch t+2 (global_prefetch_b8)
    if (tt + 2 < T) {
      const int tg2 = toff + (rev ? (T - 3 - tt) : (tt + 2));
      __builtin_prefetch(
          &xb[((size_t)(b0 + (tid >> 3)) * TTOT + tg2) * EPAD + (tid & 7) * 40], 0, 1);
    }

#pragma unroll
    for (int hg = 0; hg < 2; ++hg) {
      const int hblk = w + hg * 8;
      const size_t jrow = (size_t)(hblk * 16 + ln15);
      v8f a0i = {}, a0f = {}, a0g = {}, a0o = {};   // batch rows 0-15
      v8f a1i = {}, a1f = {}, a1g = {}, a1o = {};   // batch rows 16-31

      // ---- input projection: K over padded E ----
      const unsigned short* wib = wih_d + jrow * EPAD + bkoff;
#pragma unroll 2
      for (int kc = 0; kc < EPAD; kc += 32) {
        v16bf x0, x1;
        *(uint4*)&x0       = *(const uint4*)&xt[cur][ln15][kc + akoff];
        *((uint4*)&x0 + 1) = *(const uint4*)&xt[cur][ln15][kc + akoff + 8];
        *(uint4*)&x1       = *(const uint4*)&xt[cur][16 + ln15][kc + akoff];
        *((uint4*)&x1 + 1) = *(const uint4*)&xt[cur][16 + ln15][kc + akoff + 8];
        v16bf wI = *(const v16bf*)(wib + kc + 0 * 256 * EPAD);
        v16bf wF = *(const v16bf*)(wib + kc + 1 * 256 * EPAD);
        v16bf wG = *(const v16bf*)(wib + kc + 2 * 256 * EPAD);
        v16bf wO = *(const v16bf*)(wib + kc + 3 * 256 * EPAD);
        a0i = __builtin_amdgcn_wmma_f32_16x16x32_bf16(false, x0, false, wI, (short)0, a0i, false, false);
        a1i = __builtin_amdgcn_wmma_f32_16x16x32_bf16(false, x1, false, wI, (short)0, a1i, false, false);
        a0f = __builtin_amdgcn_wmma_f32_16x16x32_bf16(false, x0, false, wF, (short)0, a0f, false, false);
        a1f = __builtin_amdgcn_wmma_f32_16x16x32_bf16(false, x1, false, wF, (short)0, a1f, false, false);
        a0g = __builtin_amdgcn_wmma_f32_16x16x32_bf16(false, x0, false, wG, (short)0, a0g, false, false);
        a1g = __builtin_amdgcn_wmma_f32_16x16x32_bf16(false, x1, false, wG, (short)0, a1g, false, false);
        a0o = __builtin_amdgcn_wmma_f32_16x16x32_bf16(false, x0, false, wO, (short)0, a0o, false, false);
        a1o = __builtin_amdgcn_wmma_f32_16x16x32_bf16(false, x1, false, wO, (short)0, a1o, false, false);
      }
      // ---- recurrence: K over H ----
      const unsigned short* whb = whh_d + jrow * EHID + bkoff;
#pragma unroll 2
      for (int kc = 0; kc < EHID; kc += 32) {
        v16bf h0, h1;
        *(uint4*)&h0       = *(const uint4*)&ht[cur][ln15][kc + akoff];
        *((uint4*)&h0 + 1) = *(const uint4*)&ht[cur][ln15][kc + akoff + 8];
        *(uint4*)&h1       = *(const uint4*)&ht[cur][16 + ln15][kc + akoff];
        *((uint4*)&h1 + 1) = *(const uint4*)&ht[cur][16 + ln15][kc + akoff + 8];
        v16bf wI = *(const v16bf*)(whb + kc + 0 * 256 * EHID);
        v16bf wF = *(const v16bf*)(whb + kc + 1 * 256 * EHID);
        v16bf wG = *(const v16bf*)(whb + kc + 2 * 256 * EHID);
        v16bf wO = *(const v16bf*)(whb + kc + 3 * 256 * EHID);
        a0i = __builtin_amdgcn_wmma_f32_16x16x32_bf16(false, h0, false, wI, (short)0, a0i, false, false);
        a1i = __builtin_amdgcn_wmma_f32_16x16x32_bf16(false, h1, false, wI, (short)0, a1i, false, false);
        a0f = __builtin_amdgcn_wmma_f32_16x16x32_bf16(false, h0, false, wF, (short)0, a0f, false, false);
        a1f = __builtin_amdgcn_wmma_f32_16x16x32_bf16(false, h1, false, wF, (short)0, a1f, false, false);
        a0g = __builtin_amdgcn_wmma_f32_16x16x32_bf16(false, h0, false, wG, (short)0, a0g, false, false);
        a1g = __builtin_amdgcn_wmma_f32_16x16x32_bf16(false, h1, false, wG, (short)0, a1g, false, false);
        a0o = __builtin_amdgcn_wmma_f32_16x16x32_bf16(false, h0, false, wO, (short)0, a0o, false, false);
        a1o = __builtin_amdgcn_wmma_f32_16x16x32_bf16(false, h1, false, wO, (short)0, a1o, false, false);
      }

      // ---- cell update in registers; write new h (bf16) to nxt buffer ----
      const int jb = hblk * 16 + ln15;
#pragma unroll
      for (int mb = 0; mb < 2; ++mb) {
        const v8f& vi = mb ? a1i : a0i;
        const v8f& vf = mb ? a1f : a0f;
        const v8f& vg = mb ? a1g : a0g;
        const v8f& vo = mb ? a1o : a0o;
#pragma unroll
        for (int e = 0; e < 8; ++e) {
          float gi = sigm(vi[e] + bi[hg]);
          float gf = sigm(vf[e] + bf_[hg]);
          float gg = tanh_fast(vg[e] + bg_[hg]);
          float go = sigm(vo[e] + bo_[hg]);
          float cv = gf * cc[hg][mb][e] + gi * gg;
          cc[hg][mb][e] = cv;
          float hv = go * tanh_fast(cv);
          ss[hg][mb][e] += hv;
          ht[nxt][mb * 16 + rbase + e][jb] = f2bf(hv);
        }
      }
    }
  }

  // ---- write per-direction time-summed hidden states ----
#pragma unroll
  for (int hg = 0; hg < 2; ++hg) {
    const int jb = (w + hg * 8) * 16 + ln15;
#pragma unroll
    for (int mb = 0; mb < 2; ++mb) {
#pragma unroll
      for (int e = 0; e < 8; ++e) {
        hsum[(size_t)d * BATCH * EHID +
             (size_t)(b0 + mb * 16 + rbase + e) * EHID + jb] = ss[hg][mb][e];
      }
    }
  }
}

// ---------- kernel 2: per-row cosine-like scalar (compC) ----------
__global__ void rowdot_kernel(const float* __restrict__ hsum,
                              float* __restrict__ compC) {
  const int b = blockIdx.x, tid = threadIdx.x;   // 128 threads
  __shared__ float sa[128], sb[128], sc[128];
  float aa = 0, bb = 0, ab = 0;
  for (int k = tid; k < 512; k += 128) {
    int off = (k < 256) ? (b * 256 + k) : (65536 + b * 256 + (k - 256));
    float av = hsum[off];             // hA: dirs 0,1
    float bv = hsum[131072 + off];    // hB: dirs 2,3
    aa += av * av; bb += bv * bv; ab += av * bv;
  }
  sa[tid] = aa; sb[tid] = bb; sc[tid] = ab;
  __syncthreads();
  for (int s = 64; s > 0; s >>= 1) {
    if (tid < s) { sa[tid] += sa[tid + s]; sb[tid] += sb[tid + s]; sc[tid] += sc[tid + s]; }
    __syncthreads();
  }
  if (tid == 0)
    compC[b] = sc[0] / ((sqrtf(sa[0]) + 1.0f) * (sqrtf(sb[0]) + 1.0f));
}

// ---------- kernel 3: feature assembly + MLP (fp32) ----------
__global__ void __launch_bounds__(256) mlp_kernel(
    const float* __restrict__ hsum, const float* __restrict__ compC,
    const float* __restrict__ W1, const float* __restrict__ b1,
    const float* __restrict__ W2, const float* __restrict__ b2,
    float* __restrict__ out) {
  const int b = blockIdx.x, tid = threadIdx.x;
  __shared__ float ha[512], hb[512], hid[MLPN];
  __shared__ float r0[256], r1[256];
  for (int k = tid; k < 512; k += 256) {
    int off = (k < 256) ? (b * 256 + k) : (65536 + b * 256 + (k - 256));
    ha[k] = hsum[off];
    hb[k] = hsum[131072 + off];
  }
  __syncthreads();
  const float cc = compC[b];
  for (int m = tid; m < MLPN; m += 256) {
    const float* wr = W1 + (size_t)m * 1537;
    float acc = b1[m] + cc * wr[1536];
    for (int k = 0; k < 512; ++k) {
      float a = ha[k], bv = hb[k];
      acc += a * wr[k] + bv * wr[512 + k] + (a - bv) * wr[1024 + k];
    }
    hid[m] = fmaxf(acc, 0.0f);
  }
  __syncthreads();
  float p0 = 0, p1 = 0;
  for (int m = tid; m < MLPN; m += 256) {
    p0 += hid[m] * W2[m];
    p1 += hid[m] * W2[MLPN + m];
  }
  r0[tid] = p0; r1[tid] = p1;
  __syncthreads();
  for (int s = 128; s > 0; s >>= 1) {
    if (tid < s) { r0[tid] += r0[tid + s]; r1[tid] += r1[tid + s]; }
    __syncthreads();
  }
  if (tid == 0) {
    out[b * 2 + 0] = r0[0] + b2[0];
    out[b * 2 + 1] = r1[0] + b2[1];
  }
}

// ---------- launch ----------
extern "C" void kernel_launch(void* const* d_in, const int* in_sizes, int n_in,
                              void* d_out, int out_size, void* d_ws, size_t ws_size,
                              hipStream_t stream) {
  (void)in_sizes; (void)n_in; (void)out_size; (void)ws_size;
  const float* x = (const float*)d_in[0];
  Params p;
  for (int d = 0; d < 4; ++d) {
    p.wih[d] = (const float*)d_in[1 + 4 * d];
    p.whh[d] = (const float*)d_in[2 + 4 * d];
    p.bih[d] = (const float*)d_in[3 + 4 * d];
    p.bhh[d] = (const float*)d_in[4 + 4 * d];
  }
  const float* W1 = (const float*)d_in[17];
  const float* b1 = (const float*)d_in[18];
  const float* W2 = (const float*)d_in[19];
  const float* b2 = (const float*)d_in[20];

  char* ws = (char*)d_ws;
  unsigned short* wihp = (unsigned short*)ws;                          // 2,621,440 B
  unsigned short* whhb = (unsigned short*)(ws + 2621440);              // 2,097,152 B
  float* bias  = (float*)(ws + 2621440 + 2097152);                     //    16,384 B
  float* hsum  = (float*)(ws + 2621440 + 2097152 + 16384);             // 1,048,576 B
  float* compC = (float*)(ws + 2621440 + 2097152 + 16384 + 1048576);   //     1,024 B
  unsigned short* xb =
      (unsigned short*)(ws + 2621440 + 2097152 + 16384 + 1048576 + 1024); // 83,886,080 B

  const size_t convTot = XP_TOT + WIH_TOT + WHH_TOT + 4096;
  convert_kernel<<<(unsigned)((convTot + 255) / 256), 256, 0, stream>>>(
      x, p, xb, wihp, whhb, bias);
  bilstm_kernel<<<dim3(4, 8), 256, 0, stream>>>(xb, wihp, whhb, bias, hsum);
  rowdot_kernel<<<256, 128, 0, stream>>>(hsum, compC);
  mlp_kernel<<<256, 256, 0, stream>>>(hsum, compC, W1, b1, W2, b2, (float*)d_out);
}
